// Model_37520834298624
// MI455X (gfx1250) — matrix operand order repaired and checked by reading, not compiled, not run on hardware
//
#include <hip/hip_runtime.h>
#include <hip/hip_bf16.h>

typedef __attribute__((ext_vector_type(16))) __bf16 v16bf;
typedef __attribute__((ext_vector_type(8)))  __bf16 v8bf;
typedef __attribute__((ext_vector_type(8)))  float  v8f;
typedef __attribute__((ext_vector_type(4)))  unsigned int v4u;
typedef __attribute__((ext_vector_type(8)))  int v8i;
typedef __attribute__((ext_vector_type(4)))  int v4i;

#define N_NODES 100000
#define N_EDGES 300000
#define IN_DIM  29
#define K1      32            // padded layer-1 K
#define HID     128
#define NBUCKET 6
#define BROWS   64            // rows per GEMM block (buckets aligned to this)
#define KPAD    8             // extra bf16 per LDS row (16B) -> conflict-free ds_load
#define NPAD    100416        // >= N_NODES + 6*63, multiple of 64  (1569 * 64)
#define NTILES  (NPAD / BROWS)
#define CDIV(a,b) (((a)+(b)-1)/(b))

#if defined(__has_builtin)
# if __has_builtin(__builtin_amdgcn_tensor_load_to_lds)
#  define HAVE_TDM 1
# endif
#endif
#ifndef HAVE_TDM
# define HAVE_TDM 0
#endif

__device__ __forceinline__ unsigned lds_off(const void* p) {
    return (unsigned)(unsigned long long)p;   // low 32 bits of generic ptr = LDS byte offset
}

#if HAVE_TDM
// TDM: contiguous ndwords from global -> LDS, inserting 4 dwords (16B) of padding
// after every (1<<pad_interval_code) 8-byte units (i.e. per matrix row).
__device__ __forceinline__ void tdm_load_tile(unsigned lds_addr, const void* gsrc,
                                              unsigned ndwords, int pad_code) {
    unsigned long long ga = (unsigned long long)gsrc;
    v4u g0;
    g0[0] = 1u;                                                  // count=1, user descriptor
    g0[1] = lds_addr;                                            // LDS byte address
    g0[2] = (unsigned)ga;                                        // global addr [31:0]
    g0[3] = (unsigned)((ga >> 32) & 0x01FFFFFFu) | (2u << 30);   // addr [56:32] | type=2
    v8i g1;
    g1[0] = (int)((2u << 16)                                     // data_size = 4B
                | (1u << 20)                                     // pad_enable
                | ((unsigned)pad_code << 22)                     // pad_interval
                | (3u << 25));                                   // pad_amount = 4 dwords
    g1[1] = (int)((ndwords & 0xFFFFu) << 16);                    // tensor_dim0[15:0]
    g1[2] = (int)(((ndwords >> 16) & 0xFFFFu) | (1u << 16));     // tensor_dim0 hi | tensor_dim1=1
    g1[3] = (int)((ndwords & 0xFFFFu) << 16);                    // tile_dim0 = ndwords
    g1[4] = 1;                                                   // tile_dim1 = 1
    g1[5] = (int)ndwords;                                        // tensor_dim0_stride lo
    g1[6] = 0;
    g1[7] = 0;
    v4i z4 = {0, 0, 0, 0};
    v8i z8 = {0, 0, 0, 0, 0, 0, 0, 0};
    __builtin_amdgcn_tensor_load_to_lds(g0, g1, z4, z4, z8, 0);
}
#endif

// ---------------- utility kernels ----------------
__global__ void k_zero_f(float* p, long n) {
    long i = (long)blockIdx.x * blockDim.x + threadIdx.x;
    long s = (long)gridDim.x * blockDim.x;
    for (; i < n; i += s) p[i] = 0.0f;
}
__global__ void k_zero_i(int* p, long n) {
    long i = (long)blockIdx.x * blockDim.x + threadIdx.x;
    long s = (long)gridDim.x * blockDim.x;
    for (; i < n; i += s) p[i] = 0;
}
__global__ void k_cvt_f2bf(const float* __restrict__ in, __bf16* __restrict__ out, long n) {
    long i = (long)blockIdx.x * blockDim.x + threadIdx.x;
    if (i < n) out[i] = (__bf16)in[i];
}
__global__ void k_cvt_pad_w(const float* __restrict__ in, __bf16* __restrict__ out, int rows) {
    long i = (long)blockIdx.x * blockDim.x + threadIdx.x;
    long n = (long)rows * K1;
    if (i >= n) return;
    int r = (int)(i / K1), c = (int)(i % K1);
    out[i] = (c < IN_DIM) ? (__bf16)in[(long)r * IN_DIM + c] : (__bf16)0.0f;
}

// ---------------- graph preprocessing ----------------
__global__ void k_degree(const int* __restrict__ dst, int* __restrict__ deg) {
    int e = blockIdx.x * blockDim.x + threadIdx.x;
    if (e < N_EDGES) atomicAdd(&deg[dst[e]], 1);
}
__global__ void k_hist(const int* __restrict__ deg, int* __restrict__ cnt) {
    int n = blockIdx.x * blockDim.x + threadIdx.x;
    if (n >= N_NODES) return;
    int b = deg[n]; if (b > NBUCKET - 1) b = NBUCKET - 1;
    atomicAdd(&cnt[b], 1);
}
__global__ void k_offsets_tiles(const int* __restrict__ cnt, int* __restrict__ off,
                                int* __restrict__ tile_bucket) {
    __shared__ int soff[NBUCKET + 1];
    if (threadIdx.x == 0) {
        int o = 0;
        for (int d = 0; d < NBUCKET; ++d) { soff[d] = o; o += ((cnt[d] + BROWS - 1) / BROWS) * BROWS; }
        soff[NBUCKET] = o;
        for (int d = 0; d <= NBUCKET; ++d) off[d] = soff[d];
    }
    __syncthreads();
    for (int t = threadIdx.x; t < NTILES; t += blockDim.x) {
        int r = t * BROWS, b = -1;
        for (int d = 0; d < NBUCKET; ++d)
            if (r >= soff[d] && r < soff[d + 1]) b = d;
        tile_bucket[t] = b;
    }
}
__global__ void k_scatter_perm(const int* __restrict__ deg, const int* __restrict__ off,
                               int* __restrict__ cursor, int* __restrict__ perm,
                               int* __restrict__ newidx) {
    int n = blockIdx.x * blockDim.x + threadIdx.x;
    if (n >= N_NODES) return;
    int b = deg[n]; if (b > NBUCKET - 1) b = NBUCKET - 1;
    int pos = off[b] + atomicAdd(&cursor[b], 1);
    perm[pos] = n;
    newidx[n] = pos;
}
__global__ void k_remap_edges(const int* __restrict__ src, const int* __restrict__ dst,
                              const int* __restrict__ newidx,
                              int* __restrict__ srcp, int* __restrict__ dstp) {
    int e = blockIdx.x * blockDim.x + threadIdx.x;
    if (e >= N_EDGES) return;
    srcp[e] = newidx[src[e]];
    dstp[e] = newidx[dst[e]];
}
__global__ void k_permute_x(const float* __restrict__ x, const int* __restrict__ perm,
                            __bf16* __restrict__ xp) {
    long i = (long)blockIdx.x * blockDim.x + threadIdx.x;
    if (i >= (long)NPAD * K1) return;
    int row = (int)(i / K1), c = (int)(i % K1);
    int node = perm[row];
    float v = (c < IN_DIM) ? x[(long)node * IN_DIM + c] : 0.0f;
    xp[i] = (__bf16)v;
}

// ---------------- aggregation (segment_sum over dst) ----------------
__global__ void k_agg29(const int* __restrict__ src, const int* __restrict__ dstp,
                        const float* __restrict__ x, float* __restrict__ h) {
    long t = (long)blockIdx.x * blockDim.x + threadIdx.x;
    if (t >= (long)N_EDGES * 32) return;
    int e = (int)(t >> 5), f = (int)(t & 31);
    if (f >= IN_DIM) return;
    atomicAdd(&h[(long)dstp[e] * K1 + f], x[(long)src[e] * IN_DIM + f]);
}
__global__ void k_agg128(const int* __restrict__ srcp, const int* __restrict__ dstp,
                         const float* __restrict__ act, float* __restrict__ h) {
    long t = (long)blockIdx.x * blockDim.x + threadIdx.x;
    if (t >= (long)N_EDGES * HID) return;
    int e = (int)(t >> 7), f = (int)(t & 127);
    atomicAdd(&h[(long)dstp[e] * HID + f], act[(long)srcp[e] * HID + f]);
}

// ---------------- WMMA GEMM: MFConv layer ----------------
// 64 rows x 128 cols per block; TDM stages both A tiles (neighbor-sum, self) to LDS;
// 8 waves x 4 row-subtiles; weight bucket d shared by all 64 rows (alignment).
__global__ __launch_bounds__(256) void k_conv_gemm(
    const __bf16* __restrict__ Ah, const __bf16* __restrict__ Ax,
    const __bf16* __restrict__ Wl, const __bf16* __restrict__ Wr,
    const float* __restrict__ bl, const int* __restrict__ tile_bucket, int K,
    float* __restrict__ out_f, __bf16* __restrict__ out_bf, float* __restrict__ res_f) {
    __shared__ __align__(16) __bf16 smAh[BROWS * (HID + KPAD)];
    __shared__ __align__(16) __bf16 smAx[BROWS * (HID + KPAD)];
    int blk = blockIdx.x;
    int d = tile_bucket[blk];
    if (d < 0) return;
    int row0 = blk * BROWS;
    int KP = K + KPAD;
#if HAVE_TDM
    if (threadIdx.x < 32) {
        int padcode = (K == HID) ? 5 : 3;      // pad after every 64-dword / 16-dword row
        unsigned nd = (unsigned)(BROWS * K / 2);
        tdm_load_tile(lds_off(smAh), Ah + (long)row0 * K, nd, padcode);
        tdm_load_tile(lds_off(smAx), Ax + (long)row0 * K, nd, padcode);
        __builtin_amdgcn_s_wait_tensorcnt(0);
    }
#else
    for (int i = threadIdx.x; i < BROWS * K; i += 256) {
        int r = i / K, c = i % K;
        smAh[r * KP + c] = Ah[(long)row0 * K + i];
        smAx[r * KP + c] = Ax[(long)row0 * K + i];
    }
#endif
    __syncthreads();
    int wave = threadIdx.x >> 5, lane = threadIdx.x & 31;
    int col0 = wave * 16, m = lane & 15, kh = lane >> 4;
    v8f acc[4] = {};
    const __bf16* Wsel[2] = { Wl + (long)d * HID * K, Wr + (long)d * HID * K };
    const __bf16* Asel[2] = { smAh, smAx };
    for (int pass = 0; pass < 2; ++pass) {
        const __bf16* wrow = Wsel[pass] + (long)(col0 + m) * K;
        const __bf16* ab   = Asel[pass];
        __builtin_prefetch(wrow, 0, 1);
        for (int k0 = 0; k0 < K; k0 += 32) {
            v16bf b = __builtin_shufflevector(*(const v8bf*)(wrow + k0 + kh * 8), *(const v8bf*)(wrow + k0 + 16 + kh * 8), 0,1,2,3,4,5,6,7,8,9,10,11,12,13,14,15);
            #pragma unroll
            for (int s = 0; s < 4; ++s) {
                const __bf16* arow = ab + (s * 16 + m) * KP;
                v8bf alo = *(const v8bf*)(arow + k0 + kh * 8);
                v8bf ahi = *(const v8bf*)(arow + k0 + 16 + kh * 8);
                v16bf a = __builtin_shufflevector(alo, ahi,
                            0,1,2,3,4,5,6,7,8,9,10,11,12,13,14,15);
                acc[s] = __builtin_amdgcn_wmma_f32_16x16x32_bf16(
                             false, a, false, b, (short)0, acc[s], false, false);
            }
        }
    }
    float bias = bl[d * HID + col0 + m];
    #pragma unroll
    for (int s = 0; s < 4; ++s) {
        #pragma unroll
        for (int i = 0; i < 8; ++i) {
            int r = row0 + s * 16 + i + kh * 8;
            long idx = (long)r * HID + col0 + m;
            float v = acc[s][i] + bias;
            out_f[idx]  = v;
            out_bf[idx] = (__bf16)v;
            if (res_f) res_f[idx] += v;
        }
    }
}

// ---------------- WMMA GEMM: head (relu(res @ W1^T + b1)) ----------------
__global__ __launch_bounds__(256) void k_head_gemm(
    const __bf16* __restrict__ A, const __bf16* __restrict__ W1b,
    const float* __restrict__ b1, const int* __restrict__ tile_bucket,
    float* __restrict__ relu_f) {
    __shared__ __align__(16) __bf16 smA[BROWS * (HID + KPAD)];
    int blk = blockIdx.x;
    if (tile_bucket[blk] < 0) return;
    int row0 = blk * BROWS;
    const int KP = HID + KPAD;
#if HAVE_TDM
    if (threadIdx.x < 32) {
        tdm_load_tile(lds_off(smA), A + (long)row0 * HID, (unsigned)(BROWS * HID / 2), 5);
        __builtin_amdgcn_s_wait_tensorcnt(0);
    }
#else
    for (int i = threadIdx.x; i < BROWS * HID; i += 256) {
        int r = i / HID, c = i % HID;
        smA[r * KP + c] = A[(long)row0 * HID + i];
    }
#endif
    __syncthreads();
    int wave = threadIdx.x >> 5, lane = threadIdx.x & 31;
    int col0 = wave * 16, m = lane & 15, kh = lane >> 4;
    v8f acc[4] = {};
    const __bf16* wrow = W1b + (long)(col0 + m) * HID;
    __builtin_prefetch(wrow, 0, 1);
    for (int k0 = 0; k0 < HID; k0 += 32) {
        v16bf b = __builtin_shufflevector(*(const v8bf*)(wrow + k0 + kh * 8), *(const v8bf*)(wrow + k0 + 16 + kh * 8), 0,1,2,3,4,5,6,7,8,9,10,11,12,13,14,15);
        #pragma unroll
        for (int s = 0; s < 4; ++s) {
            const __bf16* arow = smA + (s * 16 + m) * KP;
            v8bf alo = *(const v8bf*)(arow + k0 + kh * 8);
            v8bf ahi = *(const v8bf*)(arow + k0 + 16 + kh * 8);
            v16bf a = __builtin_shufflevector(alo, ahi,
                        0,1,2,3,4,5,6,7,8,9,10,11,12,13,14,15);
            acc[s] = __builtin_amdgcn_wmma_f32_16x16x32_bf16(
                         false, a, false, b, (short)0, acc[s], false, false);
        }
    }
    float bias = b1[col0 + m];
    #pragma unroll
    for (int s = 0; s < 4; ++s) {
        #pragma unroll
        for (int i = 0; i < 8; ++i) {
            int r = row0 + s * 16 + i + kh * 8;
            float v = acc[s][i] + bias;
            relu_f[(long)r * HID + col0 + m] = v > 0.0f ? v : 0.0f;
        }
    }
}

// ---------------- final 128 -> 2 projection, scatter to original order ----------------
__global__ void k_head_out(const float* __restrict__ relu_f, const float* __restrict__ W2,
                           const float* __restrict__ b2, const int* __restrict__ newidx,
                           float* __restrict__ out) {
    int n = blockIdx.x * blockDim.x + threadIdx.x;
    if (n >= N_NODES) return;
    const float* r = relu_f + (long)newidx[n] * HID;
    float s0 = b2[0], s1 = b2[1];
    #pragma unroll 4
    for (int k = 0; k < HID; ++k) {
        float v = r[k];
        s0 += v * W2[k];
        s1 += v * W2[HID + k];
    }
    out[n * 2 + 0] = s0;
    out[n * 2 + 1] = s1;
}

// ---------------- host orchestration ----------------
static inline void* bump(char*& p, size_t bytes) {
    void* r = (void*)p;
    p += (bytes + 255) & ~(size_t)255;
    return r;
}

extern "C" void kernel_launch(void* const* d_in, const int* in_sizes, int n_in,
                              void* d_out, int out_size, void* d_ws, size_t ws_size,
                              hipStream_t stream) {
    (void)in_sizes; (void)n_in; (void)out_size; (void)ws_size;
    const float* x   = (const float*)d_in[0];
    const int*   ei  = (const int*)d_in[1];          // [2, E]
    const int*   src = ei;
    const int*   dst = ei + N_EDGES;
    const float* Wl1 = (const float*)d_in[2];  const float* bl1 = (const float*)d_in[3];
    const float* Wr1 = (const float*)d_in[4];
    const float* Wl2 = (const float*)d_in[5];  const float* bl2 = (const float*)d_in[6];
    const float* Wr2 = (const float*)d_in[7];
    const float* Wl3 = (const float*)d_in[8];  const float* bl3 = (const float*)d_in[9];
    const float* Wr3 = (const float*)d_in[10];
    const float* Wl4 = (const float*)d_in[11]; const float* bl4 = (const float*)d_in[12];
    const float* Wr4 = (const float*)d_in[13];
    const float* W1  = (const float*)d_in[14]; const float* b1  = (const float*)d_in[15];
    const float* W2  = (const float*)d_in[16]; const float* b2  = (const float*)d_in[17];
    float* out = (float*)d_out;

    char* w = (char*)d_ws;
    int* deg     = (int*)bump(w, (size_t)N_NODES * 4);
    int* cnt     = (int*)bump(w, 64);
    int* off     = (int*)bump(w, 64);
    int* cursor  = (int*)bump(w, 64);
    int* perm    = (int*)bump(w, (size_t)NPAD * 4);
    int* newidx  = (int*)bump(w, (size_t)N_NODES * 4);
    int* tileb   = (int*)bump(w, (size_t)NTILES * 4);
    int* srcp    = (int*)bump(w, (size_t)N_EDGES * 4);
    int* dstp    = (int*)bump(w, (size_t)N_EDGES * 4);
    __bf16* xp     = (__bf16*)bump(w, (size_t)NPAD * K1 * 2);
    float*  h1f    = (float*)bump(w, (size_t)NPAD * K1 * 4);
    __bf16* h1b    = (__bf16*)bump(w, (size_t)NPAD * K1 * 2);
    float*  hf     = (float*)bump(w, (size_t)NPAD * HID * 4);
    __bf16* hb     = (__bf16*)bump(w, (size_t)NPAD * HID * 2);
    float*  actAf  = (float*)bump(w, (size_t)NPAD * HID * 4);
    __bf16* actAb  = (__bf16*)bump(w, (size_t)NPAD * HID * 2);
    float*  actBf  = (float*)bump(w, (size_t)NPAD * HID * 4);
    __bf16* actBb  = (__bf16*)bump(w, (size_t)NPAD * HID * 2);
    float*  resf   = (float*)bump(w, (size_t)NPAD * HID * 4);
    __bf16* Wl1b = (__bf16*)bump(w, (size_t)NBUCKET * HID * K1 * 2);
    __bf16* Wr1b = (__bf16*)bump(w, (size_t)NBUCKET * HID * K1 * 2);
    __bf16* Wl2b = (__bf16*)bump(w, (size_t)NBUCKET * HID * HID * 2);
    __bf16* Wr2b = (__bf16*)bump(w, (size_t)NBUCKET * HID * HID * 2);
    __bf16* Wl3b = (__bf16*)bump(w, (size_t)NBUCKET * HID * HID * 2);
    __bf16* Wr3b = (__bf16*)bump(w, (size_t)NBUCKET * HID * HID * 2);
    __bf16* Wl4b = (__bf16*)bump(w, (size_t)NBUCKET * HID * HID * 2);
    __bf16* Wr4b = (__bf16*)bump(w, (size_t)NBUCKET * HID * HID * 2);
    __bf16* W1b  = (__bf16*)bump(w, (size_t)HID * HID * 2);

    const long nWbig = (long)NBUCKET * HID * HID;
    const long nAct  = (long)NPAD * HID;

    // --- zero state (call-idempotent; harness poisons only once) ---
    k_zero_i<<<CDIV(N_NODES, 256), 256, 0, stream>>>(deg, N_NODES);
    k_zero_i<<<1, 64, 0, stream>>>(cnt, 16);
    k_zero_i<<<1, 64, 0, stream>>>(cursor, 16);
    k_zero_i<<<CDIV(NPAD, 256), 256, 0, stream>>>(perm, NPAD);
    k_zero_f<<<CDIV((long)NPAD * K1, 256), 256, 0, stream>>>(h1f, (long)NPAD * K1);
    k_zero_f<<<CDIV(nAct, 256), 256, 0, stream>>>(resf, nAct);

    // --- degree buckets, counting sort, edge remap ---
    k_degree<<<CDIV(N_EDGES, 256), 256, 0, stream>>>(dst, deg);
    k_hist<<<CDIV(N_NODES, 256), 256, 0, stream>>>(deg, cnt);
    k_offsets_tiles<<<1, 256, 0, stream>>>(cnt, off, tileb);
    k_scatter_perm<<<CDIV(N_NODES, 256), 256, 0, stream>>>(deg, off, cursor, perm, newidx);
    k_remap_edges<<<CDIV(N_EDGES, 256), 256, 0, stream>>>(src, dst, newidx, srcp, dstp);
    k_permute_x<<<CDIV((long)NPAD * K1, 256), 256, 0, stream>>>(x, perm, xp);

    // --- weights -> bf16 ---
    k_cvt_pad_w<<<CDIV(NBUCKET * HID * K1, 256), 256, 0, stream>>>(Wl1, Wl1b, NBUCKET * HID);
    k_cvt_pad_w<<<CDIV(NBUCKET * HID * K1, 256), 256, 0, stream>>>(Wr1, Wr1b, NBUCKET * HID);
    k_cvt_f2bf<<<CDIV(nWbig, 256), 256, 0, stream>>>(Wl2, Wl2b, nWbig);
    k_cvt_f2bf<<<CDIV(nWbig, 256), 256, 0, stream>>>(Wr2, Wr2b, nWbig);
    k_cvt_f2bf<<<CDIV(nWbig, 256), 256, 0, stream>>>(Wl3, Wl3b, nWbig);
    k_cvt_f2bf<<<CDIV(nWbig, 256), 256, 0, stream>>>(Wr3, Wr3b, nWbig);
    k_cvt_f2bf<<<CDIV(nWbig, 256), 256, 0, stream>>>(Wl4, Wl4b, nWbig);
    k_cvt_f2bf<<<CDIV(nWbig, 256), 256, 0, stream>>>(Wr4, Wr4b, nWbig);
    k_cvt_f2bf<<<CDIV((long)HID * HID, 256), 256, 0, stream>>>(W1, W1b, (long)HID * HID);

    // --- layer 1 (K = 32, padded from 29) ---
    k_agg29<<<CDIV((long)N_EDGES * 32, 256), 256, 0, stream>>>(src, dstp, x, h1f);
    k_cvt_f2bf<<<CDIV((long)NPAD * K1, 256), 256, 0, stream>>>(h1f, h1b, (long)NPAD * K1);
    k_conv_gemm<<<NTILES, 256, 0, stream>>>(h1b, xp, Wl1b, Wr1b, bl1, tileb, K1,
                                            actAf, actAb, nullptr);
    // --- layer 2 ---
    k_zero_f<<<CDIV(nAct, 256), 256, 0, stream>>>(hf, nAct);
    k_agg128<<<CDIV((long)N_EDGES * HID, 256), 256, 0, stream>>>(srcp, dstp, actAf, hf);
    k_cvt_f2bf<<<CDIV(nAct, 256), 256, 0, stream>>>(hf, hb, nAct);
    k_conv_gemm<<<NTILES, 256, 0, stream>>>(hb, actAb, Wl2b, Wr2b, bl2, tileb, HID,
                                            actBf, actBb, resf);
    // --- layer 3 ---
    k_zero_f<<<CDIV(nAct, 256), 256, 0, stream>>>(hf, nAct);
    k_agg128<<<CDIV((long)N_EDGES * HID, 256), 256, 0, stream>>>(srcp, dstp, actBf, hf);
    k_cvt_f2bf<<<CDIV(nAct, 256), 256, 0, stream>>>(hf, hb, nAct);
    k_conv_gemm<<<NTILES, 256, 0, stream>>>(hb, actBb, Wl3b, Wr3b, bl3, tileb, HID,
                                            actAf, actAb, resf);
    // --- layer 4 ---
    k_zero_f<<<CDIV(nAct, 256), 256, 0, stream>>>(hf, nAct);
    k_agg128<<<CDIV((long)N_EDGES * HID, 256), 256, 0, stream>>>(srcp, dstp, actAf, hf);
    k_cvt_f2bf<<<CDIV(nAct, 256), 256, 0, stream>>>(hf, hb, nAct);
    k_conv_gemm<<<NTILES, 256, 0, stream>>>(hb, actAb, Wl4b, Wr4b, bl4, tileb, HID,
                                            actBf, actBb, resf);
    // --- head: relu(res @ W1^T + b1) @ W2^T + b2, scatter to original order ---
    k_cvt_f2bf<<<CDIV(nAct, 256), 256, 0, stream>>>(resf, hb, nAct);    // res -> bf16
    k_head_gemm<<<NTILES, 256, 0, stream>>>(hb, W1b, b1, tileb, actAf); // actAf = relu acts
    k_head_out<<<CDIV(N_NODES, 256), 256, 0, stream>>>(actAf, W2, b2, newidx, out);
}
